// CausalSelfAttention_33131377721596
// MI455X (gfx1250) — matrix operand order repaired and checked by reading, not compiled, not run on hardware
//
#include <hip/hip_runtime.h>
#include <hip/hip_bf16.h>

typedef __bf16 bf16_t;
typedef __attribute__((ext_vector_type(16))) __bf16 v16bf;
typedef __attribute__((ext_vector_type(8)))  __bf16 v8bf;
typedef __attribute__((ext_vector_type(8)))  float  v8f;
typedef __attribute__((ext_vector_type(4)))  unsigned int u32x4;
typedef __attribute__((ext_vector_type(8)))  int i32x8;
typedef __attribute__((ext_vector_type(4)))  int i32x4;

#define B_  8
#define T_  1024
#define C_  768
#define H_  12
#define D_  64
#define M_  (B_*T_)   /* 8192 rows in all GEMMs */

// ---------------------------------------------------------------------------
// Fragment load helper: two 16-byte LDS/global loads combined into one v16bf.
//   A-fragment (16x32, lane=row, half-wave splits K): hi_off = 8 elements
//   B-fragment (32x16, lane=col, 16 contiguous K):    hi_off = 8 elements
// ---------------------------------------------------------------------------
__device__ __forceinline__ v16bf ld_frag_2x16(const bf16_t* p, int hi_off) {
  v8bf lo = *(const v8bf*)p;
  v8bf hi = *(const v8bf*)(p + hi_off);
  return __builtin_shufflevector(lo, hi, 0,1,2,3,4,5,6,7,8,9,10,11,12,13,14,15);
}

__device__ __forceinline__ v8f wmma_bf16(v16bf a, v16bf b, v8f c) {
  // (neg_a, A, neg_b, B, c_mod, C, reuse_a, reuse_b)
  return __builtin_amdgcn_wmma_f32_16x16x32_bf16(false, a, false, b, (short)0, c,
                                                 false, false);
}

// Async global -> LDS copy (CDNA5, ASYNCcnt-tracked; no VGPR data path).
#define ASYNC_CP_B128(ldsoff, gptr)                                           \
  asm volatile("global_load_async_to_lds_b128 %0, %1, off"                    \
               :: "v"(ldsoff),                                                \
                  "v"((unsigned long long)(uintptr_t)(gptr))                  \
               : "memory")

// ---------------------------------------------------------------------------
// Tensor Data Mover: DMA a 2D tile (tile_d0 x tile_d1 elements of 2 bytes)
// from a row-major tensor (row length stride0 elements) into LDS, inserting
// 16B of padding after every 64B row chunk (-> LDS row stride 80B = 40 bf16).
// Wave-scalar op, TENSORcnt-tracked; issue from one wave only.
// This toolchain exposes the 6-arg builtin: (g0, g1, g2, g3, g4, cpol).
// ---------------------------------------------------------------------------
__device__ __forceinline__ void tdm_load_2d_b16(unsigned lds_off, const void* gaddr,
                                                unsigned tensor_d0, unsigned tensor_d1,
                                                unsigned tile_d0, unsigned tile_d1,
                                                unsigned stride0) {
  const unsigned long long ga = (unsigned long long)(uintptr_t)gaddr;
  u32x4 g0;
  g0[0] = 1u;                                               // count=1 (user D#)
  g0[1] = lds_off;                                          // lds_addr[31:0]
  g0[2] = (unsigned)ga;                                     // global_addr[31:0]
  g0[3] = ((unsigned)(ga >> 32) & 0x01FFFFFFu) | (2u << 30);// addr[56:32] | type=2
  i32x8 g1;
  g1[0] = (1 << 16)        // data_size = 2 bytes
        | (1 << 20)        // pad_enable
        | (3 << 22)        // pad_interval: 16 DWORDs (64B) between pads
        | (3 << 25);       // pad_amount: 4 DWORDs (16B)
  g1[1] = (int)((tensor_d0 & 0xFFFFu) << 16);                       // dim0[15:0]
  g1[2] = (int)((tensor_d0 >> 16) | ((tensor_d1 & 0xFFFFu) << 16)); // dim0 hi|dim1 lo
  g1[3] = (int)((tensor_d1 >> 16) | (tile_d0 << 16));               // dim1 hi|tile0
  g1[4] = (int)tile_d1;                                             // tile1 | tile2=0
  g1[5] = (int)stride0;                                             // dim0_stride lo
  g1[6] = 0;
  g1[7] = 0;
  const i32x4 z4 = {0, 0, 0, 0};                  // groups 2/3 unused (2D tile)
  const i32x8 z8 = {0, 0, 0, 0, 0, 0, 0, 0};
  __builtin_amdgcn_tensor_load_to_lds(g0, g1, z4, z4, z8, 0);
}

// ---------------------------------------------------------------------------
// Precision conversion kernels (run once per input; traffic trivial vs L2/HBM)
// ---------------------------------------------------------------------------
__global__ void f32_to_bf16_kernel(const float* __restrict__ in,
                                   bf16_t* __restrict__ outp, int n) {
  int i = blockIdx.x * blockDim.x + threadIdx.x;
  if (i < n) outp[i] = (bf16_t)in[i];
}

// W [K][N] fp32  ->  Wt [N][K] bf16 (so GEMM B-fragments read contiguous K)
__global__ void transpose_bf16_kernel(const float* __restrict__ in,
                                      bf16_t* __restrict__ outp, int Kd, int Nd) {
  int i = blockIdx.x * blockDim.x + threadIdx.x;
  if (i < Kd * Nd) {
    int k = i / Nd, n = i % Nd;
    outp[(size_t)n * Kd + k] = (bf16_t)in[i];
  }
}

// ---------------------------------------------------------------------------
// Tiled bf16 GEMM:  out = A[M x K] * Bt[N x K]^T + bias
// Block tile 128x128, 8 waves of 64x32, K-step 32, double-buffered LDS.
// A tile staged by the Tensor Data Mover (TENSORcnt); B tile staged by
// per-lane async global->LDS (ASYNCcnt); both overlap the WMMAs.
// mode 0: fp32 row-major [M][N]
// mode 1: bf16 -> [B,H,T,D]   (Q with scale=1/sqrt(D) folded in, and K)
// mode 2: bf16 -> [B,H,D,T]   (V transposed for the PV B-fragments)
// ---------------------------------------------------------------------------
__global__ __launch_bounds__(256)
void gemm_bf16_kernel(const bf16_t* __restrict__ A, const bf16_t* __restrict__ Bt,
                      const float* __restrict__ bias, void* __restrict__ outp,
                      int Kdim, int Ndim, int mode, float scale) {
  __shared__ bf16_t sA[2][128 * 40];  // padded stride 40 vs LDS bank conflicts
  __shared__ bf16_t sB[2][128 * 40];

  const int tid  = threadIdx.x;
  const int lane = tid & 31, wave = tid >> 5;
  const int l16  = lane & 15, hl = lane >> 4;
  const int wm = wave >> 2, wn = wave & 3;           // 2x4 wave grid, 64x32 each
  const int m0 = blockIdx.y * 128, n0 = blockIdx.x * 128;

  v8f acc[4][2] = {};

  // B staging: 256 threads x 32B cover the 128x32 bf16 tile; 2 async b128 each.
  const int srow = tid >> 1, skc = (tid & 1) * 16;
  const bf16_t* gB = Bt + (size_t)(n0 + srow) * Kdim + skc;
  unsigned ldsB[2];
  ldsB[0] = (unsigned)(uintptr_t)&sB[0][srow * 40 + skc];
  ldsB[1] = (unsigned)(uintptr_t)&sB[1][srow * 40 + skc];

  // A staging: one TDM descriptor per stage, issued by wave 0.
  const bf16_t* gA0 = A + (size_t)m0 * Kdim;
  unsigned ldsA[2];
  ldsA[0] = (unsigned)(uintptr_t)&sA[0][0];
  ldsA[1] = (unsigned)(uintptr_t)&sA[1][0];

  auto stageB = [&](int buf, int kk) {
    ASYNC_CP_B128(ldsB[buf],      gB + kk);
    ASYNC_CP_B128(ldsB[buf] + 32, gB + kk + 16);
  };
  auto stageA = [&](int buf, int kk) {
    if (wave == 0)
      tdm_load_2d_b16(ldsA[buf], gA0 + kk, (unsigned)Kdim, M_, 32u, 128u,
                      (unsigned)Kdim);
  };

  stageA(0, 0);
  stageB(0, 0);
  const int nsteps = Kdim / 32;
  for (int i = 0; i < nsteps; ++i) {
    const int cur = i & 1, nxt = cur ^ 1;
    const bool hasNext = (i + 1) < nsteps;
    if (hasNext) {
      stageA(nxt, (i + 1) * 32);                 // next-stage DMA flies under WMMAs
      stageB(nxt, (i + 1) * 32);
      asm volatile("s_wait_asynccnt 0x2" ::: "memory");   // retire older B stage
      if (wave == 0) __builtin_amdgcn_s_wait_tensorcnt(1);// retire older A stage
    } else {
      asm volatile("s_wait_asynccnt 0x0" ::: "memory");
      if (wave == 0) __builtin_amdgcn_s_wait_tensorcnt(0);
    }
    __syncthreads();                             // all waves' cur data in LDS

    const bf16_t* cA = sA[cur];
    const bf16_t* cB = sB[cur];
    v16bf a0 = ld_frag_2x16(cA + (wm * 64 +  0 + l16) * 40 + hl * 16, 8);
    v16bf a1 = ld_frag_2x16(cA + (wm * 64 + 16 + l16) * 40 + hl * 16, 8);
    v16bf a2 = ld_frag_2x16(cA + (wm * 64 + 32 + l16) * 40 + hl * 16, 8);
    v16bf a3 = ld_frag_2x16(cA + (wm * 64 + 48 + l16) * 40 + hl * 16, 8);
    v16bf b0 = ld_frag_2x16(cB + (wn * 32 +  0 + l16) * 40 + hl * 16, 8);
    v16bf b1 = ld_frag_2x16(cB + (wn * 32 + 16 + l16) * 40 + hl * 16, 8);

    acc[0][0] = wmma_bf16(a0, b0, acc[0][0]);
    acc[0][1] = wmma_bf16(a0, b1, acc[0][1]);
    acc[1][0] = wmma_bf16(a1, b0, acc[1][0]);
    acc[1][1] = wmma_bf16(a1, b1, acc[1][1]);
    acc[2][0] = wmma_bf16(a2, b0, acc[2][0]);
    acc[2][1] = wmma_bf16(a2, b1, acc[2][1]);
    acc[3][0] = wmma_bf16(a3, b0, acc[3][0]);
    acc[3][1] = wmma_bf16(a3, b1, acc[3][1]);
    __syncthreads();                             // cur fully read; safe to overwrite
  }

  #pragma unroll
  for (int fm = 0; fm < 4; ++fm) {
    #pragma unroll
    for (int fn = 0; fn < 2; ++fn) {
      const int col = n0 + wn * 32 + fn * 16 + l16;
      const float bv = bias[col];
      #pragma unroll
      for (int r = 0; r < 8; ++r) {
        const int row = m0 + wm * 64 + fm * 16 + r + 8 * hl;  // C-layout rows
        const float val = acc[fm][fn][r] + bv;
        if (mode == 0) {
          ((float*)outp)[(size_t)row * Ndim + col] = val;
        } else {
          const int bb = row >> 10, tt = row & (T_ - 1);
          const int hh = col >> 6,  dd = col & (D_ - 1);
          const bf16_t w = (bf16_t)(val * scale);
          if (mode == 1)
            ((bf16_t*)outp)[(((size_t)bb * H_ + hh) * T_ + tt) * D_ + dd] = w;
          else
            ((bf16_t*)outp)[(((size_t)bb * H_ + hh) * D_ + dd) * T_ + tt] = w;
        }
      }
    }
  }
}

// ---------------------------------------------------------------------------
// Flash attention: grid = (T/128, H, B), 8 waves/block, 16 query rows per wave.
// Q (pre-scaled by 1/sqrt(D)) kept in registers; K streamed as B-fragments
// with prefetch of the next key tile; online softmax in the C-fragment layout;
// P transposed through per-wave LDS into A-layout; PV into 4 f32 C-fragments.
// ---------------------------------------------------------------------------
__global__ __launch_bounds__(256)
void attn_kernel(const bf16_t* __restrict__ Qp, const bf16_t* __restrict__ Kp,
                 const bf16_t* __restrict__ Vt, bf16_t* __restrict__ Op) {
  __shared__ bf16_t sP[8 * 16 * 32];   // 1 KB per wave, private per wave

  const int tid = threadIdx.x, lane = tid & 31, wave = tid >> 5;
  const int l16 = lane & 15, hl = lane >> 4;
  const int qrow0 = blockIdx.x * 128 + wave * 16;
  const int h = blockIdx.y, b = blockIdx.z;
  const int bh = b * H_ + h;

  const bf16_t* Qb = Qp + (size_t)bh * T_ * D_;   // [T][D]
  const bf16_t* Kb = Kp + (size_t)bh * T_ * D_;   // [T][D]
  const bf16_t* Vb = Vt + (size_t)bh * D_ * T_;   // [D][T]
  bf16_t* sPw = sP + wave * (16 * 32);

  // Q as two A-fragments covering d=0..31 and d=32..63
  v16bf qa0 = ld_frag_2x16(Qb + (size_t)(qrow0 + l16) * D_ +  0 + hl * 16, 8);
  v16bf qa1 = ld_frag_2x16(Qb + (size_t)(qrow0 + l16) * D_ + 32 + hl * 16, 8);

  v8f o0 = {}, o1 = {}, o2 = {}, o3 = {};
  float m_r[8], l_r[8];
  #pragma unroll
  for (int r = 0; r < 8; ++r) { m_r[r] = -1e30f; l_r[r] = 0.0f; }

  for (int kt0 = 0; kt0 < qrow0 + 16; kt0 += 32) {
    // Prefetch next key tile (32 rows x 128B = one line per lane)
    if (kt0 + 32 < qrow0 + 16)
      __builtin_prefetch(Kb + (size_t)(kt0 + 32 + lane) * D_, 0, 1);

    // S = Q K^T for 32 keys: two 16x16 C-fragments, each 2 WMMAs over D=64
    v16bf kb00 = ld_frag_2x16(Kb + (size_t)(kt0      + l16) * D_ +  0 + hl * 16, 8);
    v16bf kb01 = ld_frag_2x16(Kb + (size_t)(kt0      + l16) * D_ + 32 + hl * 16, 8);
    v16bf kb10 = ld_frag_2x16(Kb + (size_t)(kt0 + 16 + l16) * D_ +  0 + hl * 16, 8);
    v16bf kb11 = ld_frag_2x16(Kb + (size_t)(kt0 + 16 + l16) * D_ + 32 + hl * 16, 8);
    v8f s0 = {}, s1 = {};
    s0 = wmma_bf16(qa0, kb00, s0);
    s0 = wmma_bf16(qa1, kb01, s0);
    s1 = wmma_bf16(qa0, kb10, s1);
    s1 = wmma_bf16(qa1, kb11, s1);

    // causal mask + online softmax (lane holds column l16, rows r + 8*hl)
    const int col0 = kt0 + l16, col1 = col0 + 16;
    float p0[8], p1[8], tmax[8], psum[8], alpha[8];
    #pragma unroll
    for (int r = 0; r < 8; ++r) {
      const int row = qrow0 + r + 8 * hl;
      p0[r] = (col0 <= row) ? s0[r] : -1e30f;
      p1[r] = (col1 <= row) ? s1[r] : -1e30f;
      tmax[r] = fmaxf(p0[r], p1[r]);
    }
    #pragma unroll
    for (int xm = 1; xm <= 8; xm <<= 1) {
      #pragma unroll
      for (int r = 0; r < 8; ++r)
        tmax[r] = fmaxf(tmax[r], __shfl_xor(tmax[r], xm, 32));
    }
    #pragma unroll
    for (int r = 0; r < 8; ++r) {
      const float mn = fmaxf(m_r[r], tmax[r]);
      alpha[r] = __expf(m_r[r] - mn);
      m_r[r] = mn;
      p0[r] = __expf(p0[r] - mn);
      p1[r] = __expf(p1[r] - mn);
      psum[r] = p0[r] + p1[r];
    }
    #pragma unroll
    for (int xm = 1; xm <= 8; xm <<= 1) {
      #pragma unroll
      for (int r = 0; r < 8; ++r)
        psum[r] += __shfl_xor(psum[r], xm, 32);
    }
    #pragma unroll
    for (int r = 0; r < 8; ++r) {
      l_r[r] = l_r[r] * alpha[r] + psum[r];
      const int prow = r + 8 * hl;
      sPw[prow * 32      + l16] = (bf16_t)p0[r];   // P tile row-major [16][32]
      sPw[prow * 32 + 16 + l16] = (bf16_t)p1[r];
      o0[r] *= alpha[r]; o1[r] *= alpha[r]; o2[r] *= alpha[r]; o3[r] *= alpha[r];
    }

    // same-wave LDS RAW: wait for the 16b stores before A-layout reload
    asm volatile("s_wait_dscnt 0x0" ::: "memory");

    v16bf pa  = ld_frag_2x16(sPw + l16 * 32 + hl * 16, 8);           // P as A-frag
    v16bf vb0 = ld_frag_2x16(Vb + (size_t)( 0 + l16) * T_ + kt0 + hl * 16, 8);
    v16bf vb1 = ld_frag_2x16(Vb + (size_t)(16 + l16) * T_ + kt0 + hl * 16, 8);
    v16bf vb2 = ld_frag_2x16(Vb + (size_t)(32 + l16) * T_ + kt0 + hl * 16, 8);
    v16bf vb3 = ld_frag_2x16(Vb + (size_t)(48 + l16) * T_ + kt0 + hl * 16, 8);
    o0 = wmma_bf16(pa, vb0, o0);
    o1 = wmma_bf16(pa, vb1, o1);
    o2 = wmma_bf16(pa, vb2, o2);
    o3 = wmma_bf16(pa, vb3, o3);
  }

  float inv[8];
  #pragma unroll
  for (int r = 0; r < 8; ++r) inv[r] = 1.0f / l_r[r];

  #pragma unroll
  for (int r = 0; r < 8; ++r) {
    const int row = qrow0 + r + 8 * hl;
    const size_t base = ((size_t)b * T_ + row) * C_ + (size_t)h * D_;
    Op[base +  0 + l16] = (bf16_t)(o0[r] * inv[r]);
    Op[base + 16 + l16] = (bf16_t)(o1[r] * inv[r]);
    Op[base + 32 + l16] = (bf16_t)(o2[r] * inv[r]);
    Op[base + 48 + l16] = (bf16_t)(o3[r] * inv[r]);
  }
}

// ---------------------------------------------------------------------------
extern "C" void kernel_launch(void* const* d_in, const int* in_sizes, int n_in,
                              void* d_out, int out_size, void* d_ws, size_t ws_size,
                              hipStream_t stream) {
  const float* x  = (const float*)d_in[0];
  const float* Wq = (const float*)d_in[1];
  const float* bq = (const float*)d_in[2];
  const float* Wk = (const float*)d_in[3];
  const float* bk = (const float*)d_in[4];
  const float* Wv = (const float*)d_in[5];
  const float* bv = (const float*)d_in[6];
  const float* Wp = (const float*)d_in[7];
  const float* bp = (const float*)d_in[8];
  float* out = (float*)d_out;

  char* ws = (char*)d_ws;
  size_t off = 0;
  auto alloc = [&](size_t bytes) -> char* {
    char* p = ws + off;
    off += (bytes + 255) & ~size_t(255);
    return p;
  };
  bf16_t* xb   = (bf16_t*)alloc((size_t)M_ * C_ * 2);  // x in bf16 (reused as attn out)
  bf16_t* Wqt  = (bf16_t*)alloc((size_t)C_ * C_ * 2);
  bf16_t* Wkt  = (bf16_t*)alloc((size_t)C_ * C_ * 2);
  bf16_t* Wvt  = (bf16_t*)alloc((size_t)C_ * C_ * 2);
  bf16_t* Wpt  = (bf16_t*)alloc((size_t)C_ * C_ * 2);
  bf16_t* Qbuf = (bf16_t*)alloc((size_t)M_ * C_ * 2);  // [B,H,T,D], pre-scaled
  bf16_t* Kbuf = (bf16_t*)alloc((size_t)M_ * C_ * 2);  // [B,H,T,D]
  bf16_t* Vtb  = (bf16_t*)alloc((size_t)M_ * C_ * 2);  // [B,H,D,T]
  bf16_t* Ab   = xb;                                   // attention output [B,T,C]

  const int nx = M_ * C_;
  f32_to_bf16_kernel<<<dim3((nx + 255) / 256), dim3(256), 0, stream>>>(x, xb, nx);
  const int nw = C_ * C_;
  transpose_bf16_kernel<<<dim3((nw + 255) / 256), dim3(256), 0, stream>>>(Wq, Wqt, C_, C_);
  transpose_bf16_kernel<<<dim3((nw + 255) / 256), dim3(256), 0, stream>>>(Wk, Wkt, C_, C_);
  transpose_bf16_kernel<<<dim3((nw + 255) / 256), dim3(256), 0, stream>>>(Wv, Wvt, C_, C_);
  transpose_bf16_kernel<<<dim3((nw + 255) / 256), dim3(256), 0, stream>>>(Wp, Wpt, C_, C_);

  dim3 ggrid(C_ / 128, M_ / 128);
  gemm_bf16_kernel<<<ggrid, dim3(256), 0, stream>>>(xb, Wqt, bq, (void*)Qbuf, C_, C_, 1, 0.125f);
  gemm_bf16_kernel<<<ggrid, dim3(256), 0, stream>>>(xb, Wkt, bk, (void*)Kbuf, C_, C_, 1, 1.0f);
  gemm_bf16_kernel<<<ggrid, dim3(256), 0, stream>>>(xb, Wvt, bv, (void*)Vtb,  C_, C_, 2, 1.0f);

  attn_kernel<<<dim3(T_ / 128, H_, B_), dim3(256), 0, stream>>>(Qbuf, Kbuf, Vtb, Ab);

  gemm_bf16_kernel<<<ggrid, dim3(256), 0, stream>>>(Ab, Wpt, bp, (void*)out, C_, C_, 0, 1.0f);
}
